// SelfAttenClassifier_54881092108989
// MI455X (gfx1250) — compile-verified, matrix operand order checked by reading, not run on hardware
//
#include <hip/hip_runtime.h>
#include <hip/hip_bf16.h>
#include <stdint.h>

// Problem constants (match reference)
#define BB 32
#define LL 2048
#define DD 256
#define CC 16
#define SCHUNK 16         // score-pass L-chunks  -> grid 32x16 = 512 WGs
#define PCHUNK 8          // pool-pass  L-chunks  -> grid 32x8  = 256 WGs
#define ECHUNK 16         // emb_sum    L-chunks  -> grid 32x16 = 512 WGs

typedef __attribute__((ext_vector_type(16))) _Float16 v16h;
typedef __attribute__((ext_vector_type(8)))  float    v8f;

// Workspace layout (floats)
#define WS_EMBSUM  0                      // B*D
#define WS_POOLACC (BB*DD)                // B*D
#define WS_DEN     (2*BB*DD)              // B
#define WS_LENS    (2*BB*DD + BB)         // B
#define WS_Y       (2*BB*DD + 2*BB)       // B*L
#define WS_ZERO_N  (2*BB*DD + 2*BB)       // everything before y needs zeroing

// ---------------------------------------------------------------------------
// Kernel 0: zero the accumulator regions of the workspace
// ---------------------------------------------------------------------------
__global__ void k_zero(float* __restrict__ p, int n) {
    int i = blockIdx.x * blockDim.x + threadIdx.x;
    if (i < n) p[i] = 0.0f;
}

// ---------------------------------------------------------------------------
// Kernel 1: emb_sum[b,:] += sum_{l in chunk} enc_w[inp[b,l],:] * mask[b,l]
//           lens[b]      += sum_{l in chunk} mask[b,l]
// grid (B, ECHUNK), block DD. Thread t owns dim t.
// ---------------------------------------------------------------------------
__global__ void k_embsum(const int* __restrict__ inp,
                         const float* __restrict__ mask,
                         const float* __restrict__ enc,
                         float* __restrict__ embsum,
                         float* __restrict__ lens) {
    const int b  = blockIdx.x;
    const int t  = threadIdx.x;
    const int l0 = blockIdx.y * (LL / ECHUNK);
    float acc = 0.0f, lenP = 0.0f;
    for (int i = 0; i < LL / ECHUNK; ++i) {
        const int l   = l0 + i;
        const int idx = inp[b * LL + l];
        const float m = mask[b * LL + l];
        if (i + 8 < LL / ECHUNK) {                 // L2 warm-up, 8 tokens ahead
            const int pidx = inp[b * LL + l + 8];
            __builtin_prefetch(enc + (size_t)pidx * DD + t, 0, 1);
        }
        acc  += enc[(size_t)idx * DD + t] * m;
        lenP += m;
    }
    atomicAdd(&embsum[b * DD + t], acc);
    if (t == 0) atomicAdd(&lens[b], lenP);
}

// ---------------------------------------------------------------------------
// Kernel 2: scores -> y_l = exp(emb_l . emb_sum) * mask_l ; partial denom.
// grid (B, SCHUNK), block 256 = 8 waves; each wave handles 16 tokens.
// emb_sum staged into LDS with the CDNA5 async global->LDS path (ASYNCcnt).
// ---------------------------------------------------------------------------
__global__ void k_scores(const int* __restrict__ inp,
                         const float* __restrict__ mask,
                         const float* __restrict__ enc,
                         const float* __restrict__ embsum,
                         float* __restrict__ yws,
                         float* __restrict__ den) {
    __shared__ float s_sum[DD];

    const int b    = blockIdx.x;
    const int t    = threadIdx.x;
    const int lane = t & 31;
    const int wave = t >> 5;
    const int l0   = blockIdx.y * (LL / SCHUNK);

    // async copy emb_sum[b,:] (1 KB) global -> LDS, one dword per lane
    {
        uint32_t ldsoff = (uint32_t)(uintptr_t)(&s_sum[t]);
        uint64_t gaddr  = (uint64_t)(uintptr_t)(embsum + b * DD + t);
        asm volatile("global_load_async_to_lds_b32 %0, %1, off"
                     :: "v"(ldsoff), "v"(gaddr) : "memory");
        asm volatile("s_wait_asynccnt 0x0" ::: "memory");
    }
    __syncthreads();

    const int d0 = lane * 8;                       // lane covers dims [d0,d0+8)
    const float4 s0 = *(const float4*)(s_sum + d0);
    const float4 s1 = *(const float4*)(s_sum + d0 + 4);

    float denP = 0.0f;
    for (int l = l0 + wave; l < l0 + LL / SCHUNK; l += 8) {
        const int   idx = inp[b * LL + l];
        const float m   = mask[b * LL + l];
        const float* row = enc + (size_t)idx * DD;
        const float4 e0 = *(const float4*)(row + d0);
        const float4 e1 = *(const float4*)(row + d0 + 4);
        float p = e0.x * s0.x + e0.y * s0.y + e0.z * s0.z + e0.w * s0.w
                + e1.x * s1.x + e1.y * s1.y + e1.z * s1.z + e1.w * s1.w;
        p *= m;                                    // emb_l = row * mask_l
        #pragma unroll
        for (int off = 16; off > 0; off >>= 1) p += __shfl_down(p, off, 32);
        if (lane == 0) {
            const float y = __expf(p) * m;         // faithful: no max-subtract
            yws[b * LL + l] = y;
            denP += y;
        }
    }
    if (lane == 0) atomicAdd(&den[b], denP);
}

// ---------------------------------------------------------------------------
// Kernel 3: poolacc[b,d] += sum_{l in chunk} enc[idx_l,d] * mask_l * y_l
// grid (B, PCHUNK), block DD. y chunk staged in LDS.
// ---------------------------------------------------------------------------
__global__ void k_pool(const int* __restrict__ inp,
                       const float* __restrict__ mask,
                       const float* __restrict__ enc,
                       const float* __restrict__ yws,
                       float* __restrict__ poolacc) {
    __shared__ float s_y[LL / PCHUNK];             // 256 floats

    const int b  = blockIdx.x;
    const int t  = threadIdx.x;
    const int l0 = blockIdx.y * (LL / PCHUNK);

    s_y[t] = yws[b * LL + l0 + t];
    __syncthreads();

    float acc = 0.0f;
    for (int i = 0; i < LL / PCHUNK; ++i) {
        const int l   = l0 + i;
        const int idx = inp[b * LL + l];
        const float m = mask[b * LL + l];
        if (i + 8 < LL / PCHUNK) {
            const int pidx = inp[b * LL + l + 8];
            __builtin_prefetch(enc + (size_t)pidx * DD + t, 0, 1);
        }
        acc += enc[(size_t)idx * DD + t] * (m * s_y[i]);
    }
    atomicAdd(&poolacc[b * DD + t], acc);
}

// ---------------------------------------------------------------------------
// Kernel 4: out[32,16] = A[32,256] @ dec_w[16,256]^T via v_wmma_f32_16x16x32_f16
// where A[m,k] = poolacc[m,k]/den[m] + embsum[m,k]/lens[m]  (folded in here).
// 1 block, 64 threads = 2 full waves; wave w computes M-tile rows [16w,16w+16).
// A-layout (16-bit A 16x32): lane<16 holds M=lane, K={0..7}+{16..23};
// lane>=16 holds M=lane-16, K={8..15}+{24..31}. B mirrors with N=lane%16.
// C/D: VGPR v -> M = v + (lane>=16 ? 8 : 0), N = lane%16.
// ---------------------------------------------------------------------------
__global__ void k_decode_wmma(const float* __restrict__ poolacc,
                              const float* __restrict__ embsum,
                              const float* __restrict__ den,
                              const float* __restrict__ lens,
                              const float* __restrict__ decw,
                              float* __restrict__ out) {
    const int lane = threadIdx.x & 31;
    const int wave = threadIdx.x >> 5;     // 0 or 1 -> M tile
    const int m0   = wave * 16;
    const int r    = lane & 15;            // M (for A) / N (for B) index
    const int hi   = (lane >> 4) & 1;      // lane group selects K sub-block

    const float idn = 1.0f / den[m0 + r];  // this lane's A-row normalizers
    const float iln = 1.0f / lens[m0 + r];

    v8f c = {};
    for (int kb = 0; kb < DD; kb += 32) {
        v16h a, bm;
        #pragma unroll
        for (int e = 0; e < 16; ++e) {
            const int p = e >> 3;          // 0 -> K base 0, 1 -> K base 16
            const int j = e & 7;
            const int k = kb + p * 16 + hi * 8 + j;
            const float av = poolacc[(m0 + r) * DD + k] * idn
                           + embsum[(m0 + r) * DD + k] * iln;
            a[e]  = (_Float16)av;
            bm[e] = (_Float16)decw[r * DD + k];   // B[k][n] = dec_w[n][k]
        }
        c = __builtin_amdgcn_wmma_f32_16x16x32_f16(
                /*neg_a=*/false, a, /*neg_b=*/false, bm,
                /*c_mod=*/(short)0, c, /*reuse_a=*/false, /*reuse_b=*/false);
    }
    #pragma unroll
    for (int v = 0; v < 8; ++v) {
        const int m = m0 + v + hi * 8;
        out[m * CC + r] = c[v];
    }
}

// ---------------------------------------------------------------------------
extern "C" void kernel_launch(void* const* d_in, const int* in_sizes, int n_in,
                              void* d_out, int out_size, void* d_ws, size_t ws_size,
                              hipStream_t stream) {
    const int*   inp  = (const int*)  d_in[0];   // [B, L] int32
    const float* mask = (const float*)d_in[1];   // [B, L] f32
    const float* enc  = (const float*)d_in[2];   // [NTOKEN+1, D] f32
    const float* decw = (const float*)d_in[3];   // [C, D] f32
    float* out = (float*)d_out;                  // [B, C] f32

    float* ws      = (float*)d_ws;
    float* embsum  = ws + WS_EMBSUM;
    float* poolacc = ws + WS_POOLACC;
    float* den     = ws + WS_DEN;
    float* lens    = ws + WS_LENS;
    float* yws     = ws + WS_Y;

    k_zero<<<(WS_ZERO_N + 255) / 256, 256, 0, stream>>>(ws, WS_ZERO_N);
    k_embsum<<<dim3(BB, ECHUNK), DD, 0, stream>>>(inp, mask, enc, embsum, lens);
    k_scores<<<dim3(BB, SCHUNK), 256, 0, stream>>>(inp, mask, enc, embsum, yws, den);
    k_pool<<<dim3(BB, PCHUNK), DD, 0, stream>>>(inp, mask, enc, yws, poolacc);
    k_decode_wmma<<<1, 64, 0, stream>>>(poolacc, embsum, den, lens, decw, out);
}
